// MuiltiHeadSelfAttention_14869176778952
// MI455X (gfx1250) — compile-verified, matrix operand order checked by reading, not run on hardware
//
#include <hip/hip_runtime.h>
#include <hip/hip_bf16.h>

#define E 512
#define H 8
#define D 64
#define LQ 1024                 // sequence length (compile-time: kills int divides)
#define MAXREL 512
#define NREL (2 * MAXREL + 1)

typedef __attribute__((ext_vector_type(16))) _Float16 v16h;
typedef __attribute__((ext_vector_type(8)))  float    v8f;

__device__ __forceinline__ v8f wmma32(v16h a, v16h b, v8f c) {
  // D = A(16x32 f16) * B(32x16 f16) + C(16x16 f32)
  return __builtin_amdgcn_wmma_f32_16x16x32_f16(
      /*neg_a=*/false, a, /*neg_b=*/false, b,
      /*c_mod=*/(short)0, c, /*reuse_a=*/false, /*reuse_b=*/false);
}

// Element e of a 16-bit A-fragment maps to k-offset within a 32-wide K-step:
// lane group 0 holds K {0..7,16..23}, group 1 holds K {8..15,24..31}.
__device__ __forceinline__ int frag_k(int e, int grp) {
  return grp * 8 + e + ((e >= 8) ? 8 : 0);
}

// Load an A-fragment (or a column-major-stored B-fragment) from a row-major
// f16 matrix [16+][ld], K-step starting at column kb.
__device__ __forceinline__ v16h frag_load_f16(const _Float16* base, int ld, int kb) {
  const int lane = threadIdx.x & 31;
  const int r = lane & 15, grp = lane >> 4;
  const _Float16* p = base + (size_t)r * ld + kb;
  v16h f;
#pragma unroll
  for (int e = 0; e < 16; ++e) f[e] = p[frag_k(e, grp)];
  return f;
}

__device__ __forceinline__ v16h frag_load_f32(const float* base, int ld, int kb) {
  const int lane = threadIdx.x & 31;
  const int r = lane & 15, grp = lane >> 4;
  const float* p = base + (size_t)r * ld + kb;
  v16h f;
#pragma unroll
  for (int e = 0; e < 16; ++e) f[e] = (_Float16)p[frag_k(e, grp)];
  return f;
}

// ---------------------------------------------------------------------------
// Tensor Data Mover: stage a dense 2D f16 tile (rows x row_elems, global row
// stride row_stride halves) into LDS.  Tracked by TENSORcnt.
// ---------------------------------------------------------------------------
#if defined(__gfx1250__) && __has_builtin(__builtin_amdgcn_tensor_load_to_lds) && \
    __has_builtin(__builtin_amdgcn_s_wait_tensorcnt)
#define USE_TDM 1
typedef unsigned int u32x4 __attribute__((ext_vector_type(4)));
typedef int          i32x4 __attribute__((ext_vector_type(4)));
typedef int          i32x8 __attribute__((ext_vector_type(8)));

__device__ __forceinline__ void tdm_load_2d_f16(const _Float16* gsrc, _Float16* lds,
                                                int rows, int row_elems, int row_stride) {
  unsigned long long ga = (unsigned long long)(uintptr_t)gsrc;   // 57-bit byte addr
  unsigned lo = (unsigned)(uintptr_t)lds;  // flat addr low 32 bits == LDS offset
  u32x4 g0;
  g0[0] = 1u;                                              // count=1 (valid), user mode
  g0[1] = lo;                                              // lds_addr (bits 63:32)
  g0[2] = (unsigned)(ga & 0xffffffffu);                    // global_addr[31:0]
  g0[3] = (unsigned)((ga >> 32) & 0x1ffffffu) | 0x80000000u; // global_addr[56:32] | type=2
  i32x8 g1;
  g1[0] = 1 << 16;                                         // data_size=1 (2 bytes), no mask
  g1[1] = (row_elems & 0xffff) << 16;                      // tensor_dim0[15:0] @ bit48
  g1[2] = ((row_elems >> 16) & 0xffff) | ((rows & 0xffff) << 16);  // dim0 hi | tensor_dim1 lo
  g1[3] = ((rows >> 16) & 0xffff) | ((row_elems & 0xffff) << 16);  // dim1 hi | tile_dim0
  g1[4] = rows & 0xffff;                                   // tile_dim1 | tile_dim2=0
  g1[5] = row_stride;                                      // tensor_dim0_stride[31:0]
  g1[6] = 0;                                               // stride hi | dim1_stride lo
  g1[7] = 0;
  i32x4 z4 = {0, 0, 0, 0};
#if __clang_major__ >= 23
  i32x8 z8 = {0, 0, 0, 0, 0, 0, 0, 0};
  __builtin_amdgcn_tensor_load_to_lds(g0, g1, z4, z4, z8, 0);
#else
  __builtin_amdgcn_tensor_load_to_lds(g0, g1, z4, z4, 0);
#endif
}
#endif

// ---------------------------------------------------------------------------
// Kernel 0: convert relative-position tables fp32 -> f16 (+ transposed relV)
// ---------------------------------------------------------------------------
__global__ void relconv_kernel(const float* __restrict__ relk,
                               const float* __restrict__ relv,
                               _Float16* __restrict__ relk16,
                               _Float16* __restrict__ relvT16) {
  int idx = blockIdx.x * blockDim.x + threadIdx.x;
  if (idx >= NREL * D) return;
  int r = idx / D, d = idx % D;
  relk16[idx] = (_Float16)relk[idx];
  relvT16[(size_t)d * NREL + r] = (_Float16)relv[idx];
}

// ---------------------------------------------------------------------------
// Kernel 1: per-head linear projection.  X:(B,L,H,D) fp32 slice of input,
// out = X @ W^T + b.   which==2 stores transposed [B,H,D,L] (for V).
// One block = 16 output rows x 64 cols; 4 waves, one 16-col subtile each.
// ---------------------------------------------------------------------------
__global__ __launch_bounds__(128)
void proj_kernel(const float* __restrict__ x, const float* __restrict__ W,
                 const float* __restrict__ bias, _Float16* __restrict__ out,
                 int which) {
  const int lane = threadIdx.x & 31;
  const int wv   = threadIdx.x >> 5;
  const int m0   = blockIdx.x * 16;      // tile over flattened (b*L + l)*H + h
  const int n0   = wv * 16;
  const int r = lane & 15, grp = lane >> 4;

  // this lane's A row
  const int flat = m0 + r;
  const int h  = flat % H;
  const int bl = flat / H;               // b*L + l
  const float* xrow = x + (size_t)bl * E + h * D;

  v8f c = {};
#pragma unroll
  for (int ks = 0; ks < 2; ++ks) {       // K = D = 64 -> 2 steps of 32
    v16h a;
#pragma unroll
    for (int e = 0; e < 16; ++e) a[e] = (_Float16)xrow[ks * 32 + frag_k(e, grp)];
    v16h bf = frag_load_f32(W + (size_t)n0 * D, D, ks * 32);  // W row-major == col-major B
    c = wmma32(a, bf, c);
  }

  const int n = n0 + r;
  const float bn = bias[n];
#pragma unroll
  for (int i = 0; i < 8; ++i) {
    int m  = grp * 8 + i;
    int fl = m0 + m;
    int hh = fl % H;                 // H, LQ compile-time -> shifts, no divides
    int bll = fl / H;
    int b = bll / LQ, l = bll % LQ;
    _Float16 val = (_Float16)(c[i] + bn);
    if (which == 2)
      out[((size_t)(b * H + hh) * D + n) * LQ + l] = val;   // V^T [B,H,D,L]
    else
      out[((size_t)(b * H + hh) * LQ + l) * D + n] = val;   // [B,H,L,D]
  }
}

// ---------------------------------------------------------------------------
// Kernel 2: flash attention with relative-position bias/value terms.
// One wave per (b, h, 16-query tile); loops over 32-key tiles.
// K and V tiles staged into LDS by the Tensor Data Mover when available.
// ---------------------------------------------------------------------------
__global__ __launch_bounds__(32)
void attn_kernel(const _Float16* __restrict__ q16, const _Float16* __restrict__ k16,
                 const _Float16* __restrict__ vT16, const int* __restrict__ mask,
                 const _Float16* __restrict__ relk16, const _Float16* __restrict__ relvT16,
                 _Float16* __restrict__ out16) {
  const int q0 = blockIdx.x * 16;
  const int h  = blockIdx.y;
  const int b  = blockIdx.z;
  const int lane = threadIdx.x;
  const int r = lane & 15, grp = lane >> 4;

  __shared__ __align__(16) _Float16 sTg[48 * 64];   // gathered relK window  [t][d]
  __shared__ __align__(16) float    sR[16 * 48];    // R = Q * Tg^T
  __shared__ __align__(16) _Float16 sP[16 * 32];    // probabilities
  __shared__ __align__(16) _Float16 sPsh[16 * 64];  // skewed probabilities (zero padded)
  __shared__ __align__(16) _Float16 sTv[64 * 64];   // gathered relV window, transposed [d][t]
#ifdef USE_TDM
  __shared__ __align__(16) _Float16 sK[32 * 64];    // TDM-staged K tile [key][d]
  __shared__ __align__(16) _Float16 sV[64 * 32];    // TDM-staged V^T tile [d][key]
#endif

  const _Float16* qbase = q16 + ((size_t)(b * H + h) * LQ + q0) * D;
  const _Float16* kbase = k16 + (size_t)(b * H + h) * LQ * D;
  const _Float16* vbase = vT16 + (size_t)(b * H + h) * D * LQ;
  const int* mrow = mask + (size_t)b * LQ;

  v16h aQ[2];
  aQ[0] = frag_load_f16(qbase, D, 0);
  aQ[1] = frag_load_f16(qbase, D, 32);

  v8f O[4] = {};
  float mrun[8], lrun[8];
#pragma unroll
  for (int i = 0; i < 8; ++i) { mrun[i] = -3.0e38f; lrun[i] = 0.0f; }

  const int NKT = LQ / 32;
  for (int kt = 0; kt < NKT; ++kt) {
    const int k0 = kt * 32;
    const int dbase = k0 - q0 - 15;   // rel distance of window slot t is dbase + t

#ifdef USE_TDM
    // kick off async DMA of the K (32x64) and V^T (64x32) tiles into LDS
    tdm_load_2d_f16(kbase + (size_t)k0 * D, sK, 32, 64, 64);
    tdm_load_2d_f16(vbase + k0, sV, 64, 32, LQ);
#endif

    // ---- gather relK window rows (48 x 64) into LDS (vectorized rows)
    for (int t = lane; t < 48; t += 32) {
      int dd = dbase + t;
      dd = dd < -MAXREL ? -MAXREL : (dd > MAXREL ? MAXREL : dd);
      const uint4* src = (const uint4*)(relk16 + (size_t)(dd + MAXREL) * D);
      uint4* dst = (uint4*)(sTg + t * 64);
#pragma unroll
      for (int j = 0; j < 8; ++j) dst[j] = src[j];
    }
    // ---- gather relV window, transposed (64 d x 64 t)
    if (dbase >= -MAXREL && dbase + 63 <= MAXREL) {
      // interior: no clamping anywhere in the window -> straight row copies
      for (int d = lane; d < 64; d += 32)
        __builtin_memcpy(sTv + d * 64,
                         relvT16 + (size_t)d * NREL + (dbase + MAXREL), 64 * sizeof(_Float16));
    } else {
      for (int idx = lane; idx < 64 * 64; idx += 32) {
        int d = idx >> 6, t = idx & 63;
        int dd = dbase + t;
        dd = dd < -MAXREL ? -MAXREL : (dd > MAXREL ? MAXREL : dd);
        sTv[d * 64 + t] = relvT16[(size_t)d * NREL + (dd + MAXREL)];
      }
    }
    // ---- zero the skewed tile
    for (int idx = lane; idx < 16 * 64; idx += 32) sPsh[idx] = (_Float16)0.0f;
    __syncthreads();
#ifdef USE_TDM
    __builtin_amdgcn_s_wait_tensorcnt((short)0);   // K/V tiles resident in LDS
#endif

    // ---- S = Q K^T (16x32) and R = Q Tg^T (16x48)
    v8f c0 = {}, c1 = {}, r0 = {}, r1 = {}, r2 = {};
#pragma unroll
    for (int ks = 0; ks < 2; ++ks) {
#ifdef USE_TDM
      v16h bK0 = frag_load_f16(sK, 64, ks * 32);
      v16h bK1 = frag_load_f16(sK + 16 * 64, 64, ks * 32);
#else
      v16h bK0 = frag_load_f16(kbase + (size_t)k0 * D, D, ks * 32);
      v16h bK1 = frag_load_f16(kbase + (size_t)(k0 + 16) * D, D, ks * 32);
#endif
      c0 = wmma32(aQ[ks], bK0, c0);
      c1 = wmma32(aQ[ks], bK1, c1);
      v16h bT0 = frag_load_f16(sTg,           64, ks * 32);
      v16h bT1 = frag_load_f16(sTg + 16 * 64, 64, ks * 32);
      v16h bT2 = frag_load_f16(sTg + 32 * 64, 64, ks * 32);
      r0 = wmma32(aQ[ks], bT0, r0);
      r1 = wmma32(aQ[ks], bT1, r1);
      r2 = wmma32(aQ[ks], bT2, r2);
    }
    // spill R to LDS for the skewed per-element lookup
#pragma unroll
    for (int i = 0; i < 8; ++i) {
      int m = grp * 8 + i;
      sR[m * 48 + r]      = r0[i];
      sR[m * 48 + 16 + r] = r1[i];
      sR[m * 48 + 32 + r] = r2[i];
    }
    __syncthreads();

    // ---- mask (before bias, as in reference), + bias, * 1/sqrt(D)
    float s0[8], s1[8];
    const int mk0 = mrow[k0 + r];
    const int mk1 = mrow[k0 + 16 + r];
#pragma unroll
    for (int i = 0; i < 8; ++i) {
      int m = grp * 8 + i;
      float v0 = c0[i], v1 = c1[i];
      if (mk0 == 0) v0 = -1.0e20f;
      if (mk1 == 0) v1 = -1.0e20f;
      v0 += sR[m * 48 + (r - m + 15)];
      v1 += sR[m * 48 + (16 + r - m + 15)];
      s0[i] = v0 * 0.125f;
      s1[i] = v1 * 0.125f;
    }

    // ---- online softmax (rows live across one 16-lane group in wave32)
#pragma unroll
    for (int i = 0; i < 8; ++i) {
      float mx = fmaxf(s0[i], s1[i]);
#pragma unroll
      for (int off = 8; off >= 1; off >>= 1) mx = fmaxf(mx, __shfl_xor(mx, off, 32));
      float newm  = fmaxf(mrun[i], mx);
      float alpha = __expf(mrun[i] - newm);
      float p0 = __expf(s0[i] - newm);
      float p1 = __expf(s1[i] - newm);
      float rs = p0 + p1;
#pragma unroll
      for (int off = 8; off >= 1; off >>= 1) rs += __shfl_xor(rs, off, 32);
      lrun[i] = lrun[i] * alpha + rs;
      mrun[i] = newm;
#pragma unroll
      for (int nt = 0; nt < 4; ++nt) O[nt][i] *= alpha;
      int m = grp * 8 + i;
      sP[m * 32 + r]        = (_Float16)p0;
      sP[m * 32 + 16 + r]   = (_Float16)p1;
      sPsh[m * 64 + (r - m + 15)]      = (_Float16)p0;  // skew: t = kj - qi + 15
      sPsh[m * 64 + (16 + r - m + 15)] = (_Float16)p1;
    }
    __syncthreads();

    // ---- O += P @ V   (V^T rows are B-operand columns)
    v16h aP = frag_load_f16(sP, 32, 0);
#pragma unroll
    for (int nt = 0; nt < 4; ++nt) {
#ifdef USE_TDM
      v16h bV = frag_load_f16(sV + nt * 16 * 32, 32, 0);
#else
      v16h bV = frag_load_f16(vbase + (size_t)(nt * 16) * LQ + k0, LQ, 0);
#endif
      O[nt] = wmma32(aP, bV, O[nt]);
    }
    // ---- O += Psh @ Tvg  (relative-value term)
    v16h aS0 = frag_load_f16(sPsh, 64, 0);
    v16h aS1 = frag_load_f16(sPsh, 64, 32);
#pragma unroll
    for (int nt = 0; nt < 4; ++nt) {
      v16h bT0 = frag_load_f16(sTv + nt * 16 * 64, 64, 0);
      v16h bT1 = frag_load_f16(sTv + nt * 16 * 64, 64, 32);
      O[nt] = wmma32(aS0, bT0, O[nt]);
      O[nt] = wmma32(aS1, bT1, O[nt]);
    }
    __syncthreads();

    if (kt + 1 < NKT)  // hint next K tile toward L2 -> global_prefetch_b8
      __builtin_prefetch(kbase + (size_t)(k0 + 32) * D + lane * 64, 0, 3);
  }

  // ---- finalize: divide by row sums, store f16 [B,L,H,D] for the FC GEMM
#pragma unroll
  for (int i = 0; i < 8; ++i) {
    int m = grp * 8 + i;
    float inv = 1.0f / lrun[i];
#pragma unroll
    for (int nt = 0; nt < 4; ++nt) {
      int d = nt * 16 + r;
      out16[((size_t)(b * LQ + (q0 + m)) * H + h) * D + d] = (_Float16)(O[nt][i] * inv);
    }
  }
}

// ---------------------------------------------------------------------------
// Kernel 3: final FC.  out(B*L,512) = X(B*L,512) @ Wfc^T + bfc, fp32 output.
// One wave per 16x32 output tile.
// ---------------------------------------------------------------------------
__global__ __launch_bounds__(32)
void fc_kernel(const _Float16* __restrict__ x16, const float* __restrict__ W,
               const float* __restrict__ bias, float* __restrict__ out) {
  const int m0 = blockIdx.x * 16;
  const int n0 = blockIdx.y * 32;
  const int lane = threadIdx.x;
  const int r = lane & 15, grp = lane >> 4;

  v8f c0 = {}, c1 = {};
  for (int k = 0; k < E; k += 32) {
    v16h a  = frag_load_f16(x16 + (size_t)m0 * E, E, k);
    v16h b0 = frag_load_f32(W + (size_t)n0 * E, E, k);        // Wfc rows == B columns
    v16h b1 = frag_load_f32(W + (size_t)(n0 + 16) * E, E, k);
    c0 = wmma32(a, b0, c0);
    c1 = wmma32(a, b1, c1);
  }
  const float bn0 = bias[n0 + r], bn1 = bias[n0 + 16 + r];
#pragma unroll
  for (int i = 0; i < 8; ++i) {
    int m = m0 + grp * 8 + i;
    out[(size_t)m * E + n0 + r]      = c0[i] + bn0;
    out[(size_t)m * E + n0 + 16 + r] = c1[i] + bn1;
  }
}

// ---------------------------------------------------------------------------
extern "C" void kernel_launch(void* const* d_in, const int* in_sizes, int n_in,
                              void* d_out, int out_size, void* d_ws, size_t ws_size,
                              hipStream_t stream) {
  const float* query = (const float*)d_in[0];
  const float* key   = (const float*)d_in[1];
  const float* value = (const float*)d_in[2];
  const int*   mask  = (const int*)d_in[3];
  const float* Wq  = (const float*)d_in[4];
  const float* bq  = (const float*)d_in[5];
  const float* Wk  = (const float*)d_in[6];
  const float* bk  = (const float*)d_in[7];
  const float* Wv  = (const float*)d_in[8];
  const float* bv  = (const float*)d_in[9];
  const float* Wfc = (const float*)d_in[10];
  const float* bfc = (const float*)d_in[11];
  const float* relk = (const float*)d_in[12];
  const float* relv = (const float*)d_in[13];

  const int B  = in_sizes[0] / (LQ * E);
  const int BL = B * LQ;

  // workspace carve-up (all f16 tensors)
  char* ws = (char*)d_ws;
  size_t off = 0;
  auto carve = [&](size_t bytes) {
    char* p = ws + off;
    off = (off + bytes + 255) & ~(size_t)255;
    return p;
  };
  _Float16* q16     = (_Float16*)carve((size_t)BL * E * sizeof(_Float16)); // [B,H,L,D]
  _Float16* k16     = (_Float16*)carve((size_t)BL * E * sizeof(_Float16)); // [B,H,L,D]
  _Float16* vT16    = (_Float16*)carve((size_t)BL * E * sizeof(_Float16)); // [B,H,D,L]
  _Float16* attn16  = (_Float16*)carve((size_t)BL * E * sizeof(_Float16)); // [B,L,E]
  _Float16* relk16  = (_Float16*)carve((size_t)NREL * D * sizeof(_Float16));
  _Float16* relvT16 = (_Float16*)carve((size_t)NREL * D * sizeof(_Float16));
  (void)ws_size; (void)n_in; (void)out_size;

  // 0) table conversion
  relconv_kernel<<<(NREL * D + 255) / 256, 256, 0, stream>>>(relk, relv, relk16, relvT16);

  // 1) Q/K/V projections (per-head 64x64 GEMMs, f16 WMMA)
  const int ptiles = (BL * H) / 16;
  proj_kernel<<<ptiles, 128, 0, stream>>>(query, Wq, bq, q16, 0);
  proj_kernel<<<ptiles, 128, 0, stream>>>(key,   Wk, bk, k16, 1);
  proj_kernel<<<ptiles, 128, 0, stream>>>(value, Wv, bv, vT16, 2);

  // 2) flash attention with relative bias/value terms (TDM-staged K/V tiles)
  attn_kernel<<<dim3(LQ / 16, H, B), 32, 0, stream>>>(q16, k16, vT16, mask,
                                                      relk16, relvT16, attn16);

  // 3) final FC, fp32 output
  fc_kernel<<<dim3(BL / 16, E / 32), 32, 0, stream>>>(attn16, Wfc, bfc, (float*)d_out);
}
